// AttentionHead_40286793236700
// MI455X (gfx1250) — matrix operand order repaired and checked by reading, not compiled, not run on hardware
//
#include <hip/hip_runtime.h>

// ---------------------------------------------------------------------------
// Linear attention (elu+1 feature map) for B=16, S=1024, DM=DH=1024 on gfx1250.
// Matmuls: v_wmma_f32_16x16x32_bf16 (wave32 WMMA), f32 accumulation.
// kv/out GEMMs stage tiles with the Tensor Data Mover (tensor_load_to_lds,
// TENSORcnt) into double-buffered LDS; QKV projection converts f32->bf16
// while staging manually.
// ---------------------------------------------------------------------------

typedef __bf16 bf16;
typedef __attribute__((ext_vector_type(16))) __bf16 v16bf;
typedef __attribute__((ext_vector_type(8)))  float  v8f;
typedef __attribute__((ext_vector_type(4)))  unsigned int v4u;
typedef __attribute__((ext_vector_type(8)))  int v8i;
typedef __attribute__((ext_vector_type(4)))  int v4i;

constexpr int Bn = 16;    // batch
constexpr int Sn = 1024;  // sequence
constexpr int Dm = 1024;  // model dim
constexpr int Dh = 1024;  // head dim (== Sn, required by reference)

// GEMM tiling: block 128x128x32, 256 threads = 8 wave32s (2x4 wave grid),
// each wave owns a 64x32 tile = 4x2 WMMA 16x16 accumulators.
constexpr int BM = 128, BN = 128, BK = 32;
constexpr int ASLD = 48;   // LDS row stride (bf16) for 32-wide tiles  (32 + 8dw pad)
constexpr int BSLD = 136;  // LDS row stride (bf16) for 128-wide tiles (128 + 4dw pad)

// ---------------- Tensor Data Mover: 2D tile load into LDS ----------------
// Builds the D# (groups 0/1 per CDNA5 ISA) and issues
// TENSOR_LOAD_TO_LDS. data_size=1 (2-byte elems). LDS padding via
// pad_interval/pad_amount so tiles land with ASLD/BSLD row strides.
__device__ __forceinline__ void tdm_load_2d(unsigned lds_byte_addr, const void* gptr,
                                            unsigned tile_x_elems, unsigned tile_y_rows,
                                            unsigned tensor_dim0, unsigned tensor_dim1,
                                            unsigned row_stride_elems,
                                            unsigned pad_interval_code,
                                            unsigned pad_amount_code) {
  const unsigned long long ga = (unsigned long long)(size_t)gptr;
  v4u g0;
  g0[0] = 1u;                                            // count=1, user descriptor
  g0[1] = lds_byte_addr;                                 // lds_addr [63:32]
  g0[2] = (unsigned)(ga & 0xFFFFFFFFu);                  // global_addr lo
  g0[3] = (unsigned)((ga >> 32) & 0x1FFFFFFu) | (2u << 30);  // addr hi | type=2

  unsigned w[8] = {0u, 0u, 0u, 0u, 0u, 0u, 0u, 0u};
  w[0] = (1u << 16)                      // data_size = 1 -> 2 bytes
       | (1u << 20)                      // pad_enable (LDS dest padding)
       | (pad_interval_code << 22)       // dwords between pads
       | (pad_amount_code << 25);        // pad dwords
  // tensor_dim0: bits 79:48
  w[1] |= (tensor_dim0 & 0xFFFFu) << 16;
  w[2] |= tensor_dim0 >> 16;
  // tensor_dim1: bits 111:80
  w[2] |= (tensor_dim1 & 0xFFFFu) << 16;
  w[3] |= (tensor_dim1 >> 16) & 0xFFFFu;
  // tile_dim0: bits 127:112 ; tile_dim1: bits 143:128
  w[3] |= tile_x_elems << 16;
  w[4] |= tile_y_rows & 0xFFFFu;
  // tensor_dim0_stride: bits 207:160 (48-bit, ours fit in 32)
  w[5] = row_stride_elems;

  v8i g1;
#pragma unroll
  for (int i = 0; i < 8; ++i) g1[i] = (int)w[i];
  v4i z4 = {0, 0, 0, 0};
#if __clang_major__ >= 23
  v8i z8 = {0, 0, 0, 0, 0, 0, 0, 0};
  __builtin_amdgcn_tensor_load_to_lds(g0, g1, z4, z4, z8, 0);
#else
  __builtin_amdgcn_tensor_load_to_lds(g0, g1, z4, z4, 0);
#endif
}

// ---------------- Manual staging (f32 source, cvt to bf16) ----------------
__device__ __forceinline__ void copy_rows_f32(bf16* dst, int dld, const float* src,
                                              int sld, int rows, int relems, int tid) {
  const int qper = relems >> 2;           // float4 per row
  const int total = rows * qper;
  for (int i = tid; i < total; i += 256) {
    const int r = i / qper, q = i - r * qper;
    const float4 f = *(reinterpret_cast<const float4*>(src + (size_t)r * sld) + q);
    bf16* d = dst + r * dld + q * 4;
    d[0] = (bf16)f.x; d[1] = (bf16)f.y; d[2] = (bf16)f.z; d[3] = (bf16)f.w;
  }
}

// ---------------- WMMA fragment loads (wave32 layouts, ISA 7.12.2) ----------
__device__ __forceinline__ v16bf load_afrag(const bf16* As, int ld, int mbase, int lane) {
  const int half = lane >> 4;
  const bf16* row = As + (mbase + (lane & 15)) * ld;
  v16bf a;
#pragma unroll
  for (int e = 0; e < 16; ++e) {
    const int k = half * 16 + e;
    a[e] = row[k];
  }
  return a;
}

__device__ __forceinline__ v16bf load_bfrag_nmajor(const bf16* Bs, int ld, int nbase, int lane) {
  const int kbase = (lane >> 4) * 16;
  const bf16* col = Bs + (nbase + (lane & 15)) * ld;
  v16bf b;
#pragma unroll
  for (int e = 0; e < 16; ++e) b[e] = col[kbase + e];
  return b;
}

__device__ __forceinline__ v16bf load_bfrag_kmajor(const bf16* Bs, int ld, int nbase, int lane) {
  const int kbase = (lane >> 4) * 16;
  const int n = nbase + (lane & 15);
  v16bf b;
#pragma unroll
  for (int e = 0; e < 16; ++e) b[e] = Bs[(kbase + e) * ld + n];
  return b;
}

// ---------------- WMMA tile compute on one staged LDS buffer ----------------
template <bool BT>
__device__ __forceinline__ void compute_tiles(const bf16* As, const bf16* Bs,
                                              v8f (&acc)[4][2], int lane, int wm, int wn) {
  v16bf af[4], bfv[2];
#pragma unroll
  for (int mi = 0; mi < 4; ++mi) af[mi] = load_afrag(As, ASLD, wm + mi * 16, lane);
#pragma unroll
  for (int ni = 0; ni < 2; ++ni)
    bfv[ni] = BT ? load_bfrag_nmajor(Bs, ASLD, wn + ni * 16, lane)
                 : load_bfrag_kmajor(Bs, BSLD, wn + ni * 16, lane);
#pragma unroll
  for (int mi = 0; mi < 4; ++mi)
#pragma unroll
    for (int ni = 0; ni < 2; ++ni)
      acc[mi][ni] = __builtin_amdgcn_wmma_f32_16x16x32_bf16(
          false, af[mi], false, bfv[ni], (short)0, acc[mi][ni], false, false);
}

// ---------------- TDM double-buffered GEMM main loop (bf16 sources) ---------
// BT=true : LDS B tile [BN][ASLD]; Bg at (n0, k=0), ldb = K stride
// BT=false: LDS B tile [BK][BSLD]; Bg at (k=0, n0), ldb = N stride
template <bool BT>
__device__ __forceinline__ void gemm_mainloop_tdm(const bf16* __restrict__ Ag, int lda,
                                                  const bf16* __restrict__ Bg, int ldb, int K,
                                                  bf16* As0, bf16* As1, bf16* Bs0, bf16* Bs1,
                                                  v8f (&acc)[4][2]) {
  const int tid = threadIdx.x;
  const int lane = tid & 31;
  const int wave = tid >> 5;
  const int wm = (wave >> 2) * 64, wn = (wave & 3) * 32;
  const bool issuer = (wave == 0);   // one wave drives the TDM (EXEC ignored)

  bf16* AsBuf[2] = {As0, As1};
  bf16* BsBuf[2] = {Bs0, Bs1};

  auto stage = [&](int buf, int kt) {
    // A tile: BM rows x BK elems -> [BM][ASLD]: pad_interval=16dw(code 3), pad=8dw(code 7)
    tdm_load_2d((unsigned)(size_t)AsBuf[buf], Ag + kt, BK, BM,
                (unsigned)lda, (unsigned)Sn, (unsigned)lda, 3u, 7u);
    if (BT)   // B^T tile: BN rows x BK elems -> [BN][ASLD]
      tdm_load_2d((unsigned)(size_t)BsBuf[buf], Bg + kt, BK, BN,
                  (unsigned)ldb, (unsigned)Sn, (unsigned)ldb, 3u, 7u);
    else      // B tile: BK rows x BN elems -> [BK][BSLD]: interval=64dw(5), pad=4dw(3)
      tdm_load_2d((unsigned)(size_t)BsBuf[buf], Bg + (size_t)kt * ldb, BN, BK,
                  (unsigned)ldb, (unsigned)Sn, (unsigned)ldb, 5u, 3u);
  };

  const int nIter = K / BK;
  if (issuer) stage(0, 0);
  for (int it = 0; it < nIter; ++it) {
    const int cur = it & 1;
    if (issuer) __builtin_amdgcn_s_wait_tensorcnt(0);  // cur buffer landed in LDS
    __syncthreads();                                   // (also fences prior reads of cur^1)
    if (issuer && it + 1 < nIter) stage(cur ^ 1, (it + 1) * BK);
    compute_tiles<BT>(AsBuf[cur], BsBuf[cur], acc, lane, wm, wn);
  }
}

// ---------------- Manual GEMM main loop (f32 sources, QKV projection) -------
__device__ __forceinline__ void gemm_mainloop_f32(const float* __restrict__ Ag, int lda,
                                                  const float* __restrict__ Bg, int ldb, int K,
                                                  bf16* As, bf16* Bs, v8f (&acc)[4][2]) {
  const int tid = threadIdx.x;
  const int lane = tid & 31;
  const int wave = tid >> 5;
  const int wm = (wave >> 2) * 64, wn = (wave & 3) * 32;

  for (int kt = 0; kt < K; kt += BK) {
    __syncthreads();
    copy_rows_f32(As, ASLD, Ag + kt, lda, BM, BK, tid);
    copy_rows_f32(Bs, ASLD, Bg + kt, ldb, BN, BK, tid);
    if (tid < BM && kt + BK < K)   // hint next A tile toward the caches
      __builtin_prefetch(Ag + (size_t)tid * lda + kt + BK, 0, 1);
    __syncthreads();
    compute_tiles<true>(As, Bs, acc, lane, wm, wn);
  }
}

// ---------------- Kernel 1: fused QKV projection + elu+1 + mask -------------
__global__ __launch_bounds__(256) void qkv_proj_kernel(
    const float* __restrict__ x,
    const float* __restrict__ Wq, const float* __restrict__ Wk, const float* __restrict__ Wv,
    const float* __restrict__ bq, const float* __restrict__ bk, const float* __restrict__ bv,
    const int* __restrict__ mask,
    bf16* __restrict__ phiq, bf16* __restrict__ phik, bf16* __restrict__ vb) {
  __shared__ __align__(16) bf16 As[BM * ASLD];
  __shared__ __align__(16) bf16 Bs[BN * ASLD];

  const int which = blockIdx.z;
  const float* W    = which == 0 ? Wq : which == 1 ? Wk : Wv;
  const float* bias = which == 0 ? bq : which == 1 ? bk : bv;
  bf16* out         = which == 0 ? phiq : which == 1 ? phik : vb;

  const int m0 = blockIdx.y * BM;   // row in flattened (B*S)
  const int n0 = blockIdx.x * BN;   // output feature d

  v8f acc[4][2] = {};
  gemm_mainloop_f32(x + (size_t)m0 * Dm, Dm, W + (size_t)n0 * Dm, Dm, Dm, As, Bs, acc);

  const int lane = threadIdx.x & 31, wave = threadIdx.x >> 5;
  const int wm = (wave >> 2) * 64, wn = (wave & 3) * 32;
#pragma unroll
  for (int mi = 0; mi < 4; ++mi)
#pragma unroll
    for (int ni = 0; ni < 2; ++ni)
#pragma unroll
      for (int r = 0; r < 8; ++r) {
        const int m = m0 + wm + mi * 16 + (lane >> 4) * 8 + r;
        const int n = n0 + wn + ni * 16 + (lane & 15);
        float val = acc[mi][ni][r] + bias[n];
        if (which < 2) val = (val > 0.0f) ? val + 1.0f : __expf(val);  // elu(x)+1
        if (mask[m] == 1) val = 0.0f;                                  // padded row
        out[(size_t)m * Dh + n] = (bf16)val;
      }
}

// ---------------- Kernel 2: k_sum[b,d] = sum_t phi_k[b,t,d] -----------------
__global__ __launch_bounds__(256) void ksum_kernel(const bf16* __restrict__ phik,
                                                   float* __restrict__ ksum) {
  const int d = blockIdx.x * 256 + threadIdx.x;
  const int b = blockIdx.y;
  const bf16* p = phik + (size_t)b * Sn * Dh + d;
  float s = 0.0f;
  for (int t = 0; t < Sn; ++t) s += (float)p[(size_t)t * Dh];
  ksum[b * Dh + d] = s;
}

// ---------------- Kernel 3: z[b,s] = 1/max(phi_q[b,s,:].k_sum[b,:], eps) ----
__global__ __launch_bounds__(256) void denom_kernel(const bf16* __restrict__ phiq,
                                                    const float* __restrict__ ksum,
                                                    float* __restrict__ zrow) {
  const int row  = (blockIdx.x * 256 + threadIdx.x) >> 5;  // one wave per (b,s)
  const int lane = threadIdx.x & 31;
  const int b = row >> 10;
  const bf16* q = phiq + (size_t)row * Dh;
  const float* ks = ksum + b * Dh;
  float s = 0.0f;
  for (int d = lane; d < Dh; d += 32) s += (float)q[d] * ks[d];
#pragma unroll
  for (int off = 16; off > 0; off >>= 1) s += __shfl_down(s, off, 32);
  if (lane == 0) zrow[row] = 1.0f / fmaxf(s, 1e-6f);
}

// ---------------- Kernel 4: kv[b,s,t] = sum_d v[b,s,d]*phi_k[b,t,d] ---------
__global__ __launch_bounds__(256) void kv_gemm_kernel(const bf16* __restrict__ vb,
                                                      const bf16* __restrict__ phik,
                                                      bf16* __restrict__ kvb) {
  __shared__ __align__(16) bf16 As0[BM * ASLD], As1[BM * ASLD];
  __shared__ __align__(16) bf16 Bs0[BN * ASLD], Bs1[BN * ASLD];
  const int b = blockIdx.z;
  const int m0 = blockIdx.y * BM;   // s
  const int n0 = blockIdx.x * BN;   // t
  const bf16* A  = vb   + (size_t)b * Sn * Dh + (size_t)m0 * Dh;
  const bf16* Bm = phik + (size_t)b * Sn * Dh + (size_t)n0 * Dh;  // row t contiguous in d
  v8f acc[4][2] = {};
  gemm_mainloop_tdm<true>(A, Dh, Bm, Dh, Dh, As0, As1, Bs0, Bs1, acc);

  const int lane = threadIdx.x & 31, wave = threadIdx.x >> 5;
  const int wm = (wave >> 2) * 64, wn = (wave & 3) * 32;
  bf16* outp = kvb + (size_t)b * Sn * Sn;
#pragma unroll
  for (int mi = 0; mi < 4; ++mi)
#pragma unroll
    for (int ni = 0; ni < 2; ++ni)
#pragma unroll
      for (int r = 0; r < 8; ++r) {
        const int m = m0 + wm + mi * 16 + (lane >> 4) * 8 + r;
        const int n = n0 + wn + ni * 16 + (lane & 15);
        outp[(size_t)m * Sn + n] = (bf16)acc[mi][ni][r];
      }
}

// ---------------- Kernel 5: out[b,s,t] = (phi_q @ kv)[s,t] * z[b,s] ---------
__global__ __launch_bounds__(256) void out_gemm_kernel(const bf16* __restrict__ phiq,
                                                       const bf16* __restrict__ kvb,
                                                       const float* __restrict__ zrow,
                                                       float* __restrict__ out) {
  __shared__ __align__(16) bf16 As0[BM * ASLD], As1[BM * ASLD];
  __shared__ __align__(16) bf16 Bs0[BK * BSLD], Bs1[BK * BSLD];
  const int b = blockIdx.z;
  const int m0 = blockIdx.y * BM;   // s
  const int n0 = blockIdx.x * BN;   // t
  const bf16* A  = phiq + (size_t)b * Sn * Dh + (size_t)m0 * Dh;
  const bf16* Bm = kvb  + (size_t)b * Sn * Sn + n0;   // kv row d contiguous in t
  v8f acc[4][2] = {};
  gemm_mainloop_tdm<false>(A, Dh, Bm, Sn, Dh, As0, As1, Bs0, Bs1, acc);

  const int lane = threadIdx.x & 31, wave = threadIdx.x >> 5;
  const int wm = (wave >> 2) * 64, wn = (wave & 3) * 32;
  float* outp = out + (size_t)b * Sn * Sn;
  const float* zp = zrow + b * Sn;
#pragma unroll
  for (int mi = 0; mi < 4; ++mi)
#pragma unroll
    for (int ni = 0; ni < 2; ++ni) {
      const int mrow = wm + mi * 16 + (lane >> 4) * 8;
#pragma unroll
      for (int r = 0; r < 8; ++r) {
        const int m = m0 + mrow + r;
        const int n = n0 + wn + ni * 16 + (lane & 15);
        outp[(size_t)m * Sn + n] = acc[mi][ni][r] * zp[m];
      }
    }
}

// ---------------- Host launcher ----------------
extern "C" void kernel_launch(void* const* d_in, const int* in_sizes, int n_in,
                              void* d_out, int out_size, void* d_ws, size_t ws_size,
                              hipStream_t stream) {
  (void)in_sizes; (void)n_in; (void)out_size; (void)ws_size;
  const float* x    = (const float*)d_in[0];
  const int*   mask = (const int*)d_in[1];
  const float* Wq   = (const float*)d_in[2];
  const float* bq   = (const float*)d_in[3];
  const float* Wk   = (const float*)d_in[4];
  const float* bk   = (const float*)d_in[5];
  const float* Wv   = (const float*)d_in[6];
  const float* bv   = (const float*)d_in[7];
  float* out = (float*)d_out;

  // Workspace carve (256B aligned): phi_q, phi_k, v (bf16), kv (bf16), k_sum, z
  char* w = (char*)d_ws;
  auto carve = [&](size_t bytes) {
    char* p = w;
    w += (bytes + 255) & ~(size_t)255;
    return p;
  };
  bf16*  phiq = (bf16*)carve((size_t)Bn * Sn * Dh * sizeof(bf16));
  bf16*  phik = (bf16*)carve((size_t)Bn * Sn * Dh * sizeof(bf16));
  bf16*  vb   = (bf16*)carve((size_t)Bn * Sn * Dh * sizeof(bf16));
  bf16*  kvb  = (bf16*)carve((size_t)Bn * Sn * Sn * sizeof(bf16));
  float* ksum = (float*)carve((size_t)Bn * Dh * sizeof(float));
  float* zrow = (float*)carve((size_t)Bn * Sn * sizeof(float));

  const dim3 blk(256);
  qkv_proj_kernel<<<dim3(Dh / BN, (Bn * Sn) / BM, 3), blk, 0, stream>>>(
      x, Wq, Wk, Wv, bq, bk, bv, mask, phiq, phik, vb);
  ksum_kernel<<<dim3(Dh / 256, Bn), blk, 0, stream>>>(phik, ksum);
  denom_kernel<<<dim3((Bn * Sn * 32) / 256), blk, 0, stream>>>(phiq, ksum, zrow);
  kv_gemm_kernel<<<dim3(Sn / BN, Sn / BM, Bn), blk, 0, stream>>>(vb, phik, kvb);
  out_gemm_kernel<<<dim3(Sn / BN, Sn / BM, Bn), blk, 0, stream>>>(phiq, kvb, zrow, out);
}
